// XposMultiheadAttention_64106681860433
// MI455X (gfx1250) — compile-verified, matrix operand order checked by reading, not run on hardware
//
#include <hip/hip_runtime.h>
#include <hip/hip_bf16.h>

typedef __bf16 bf16_t;
typedef __attribute__((ext_vector_type(16))) __bf16 v16bf;
typedef __attribute__((ext_vector_type(8)))  __bf16 v8bf;
typedef __attribute__((ext_vector_type(8)))  float  v8f;
typedef __attribute__((ext_vector_type(4))) unsigned int u32x4;
typedef __attribute__((ext_vector_type(4))) int i32x4;
typedef __attribute__((ext_vector_type(8))) int i32x8;

#define EMBED 1024
#define HEADS 16
#define HDIM  64
#define SEQ   2048
#define BATCH 2
#define MTOT  (BATCH * SEQ)     /* 4096 rows in flattened (b,t) */
#define NBH   (BATCH * HEADS)   /* 32 head-batches */

// ---------------------------------------------------------------------------
// WMMA helper: D = A(16x32 bf16) * B(32x16 bf16) + C(16x16 f32)
// ---------------------------------------------------------------------------
__device__ __forceinline__ v8f wmma_bf16(v16bf a, v16bf b, v8f c) {
  return __builtin_amdgcn_wmma_f32_16x16x32_bf16(
      /*neg_a=*/false, a, /*neg_b=*/false, b,
      /*c_mod=*/(short)0, c, /*reuse_a=*/false, /*reuse_b=*/false);
}

// Load a 16x32 bf16 fragment where each lane owns one *row* of the operand.
// Per ISA layout: lanes 0-15 hold K=0..7 (f[0..7]) and K=16..23 (f[8..15]);
// lanes 16-31 hold K=8..15 and K=24..31.
__device__ __forceinline__ v16bf load_frag_row(const bf16_t* base, int ld, int k0) {
  const int lane = threadIdx.x & 31;
  const int r  = lane & 15;
  const int hi = lane >> 4;
  const bf16_t* p = base + (size_t)r * ld + k0 + hi * 8;
  v8bf lo = *(const v8bf*)(p);
  v8bf hb = *(const v8bf*)(p + 16);
  v16bf f;
#pragma unroll
  for (int i = 0; i < 8; ++i) { f[i] = lo[i]; f[i + 8] = hb[i]; }
  return f;
}

// ---------------------------------------------------------------------------
// Tensor Data Mover: async-load a [64 x 64] bf16 tile (rows contiguous,
// row stride 64 elements) from global into LDS.  D# built per ISA ch.8.
// ---------------------------------------------------------------------------
__device__ __forceinline__ void tdm_load_k64x64(unsigned lds_addr,
                                                unsigned long long gaddr) {
  u32x4 g0;
  g0[0] = 1u;                                           // count=1, user mode
  g0[1] = lds_addr;                                     // lds_addr
  g0[2] = (unsigned)(gaddr & 0xFFFFFFFFull);            // global_addr[31:0]
  g0[3] = (unsigned)((gaddr >> 32) & 0x1FFFFFFull)      // global_addr[56:32]
          | (2u << 30);                                 // type=2 ("image")
  i32x8 g1;
  g1[0] = 0x00010000;                                   // data_size=1 (2B)
  g1[1] = (int)(64u << 16);                             // tensor_dim0 = 64
  g1[2] = (int)(2048u << 16);                           // tensor_dim1 = 2048
  g1[3] = (int)(64u << 16);                             // tile_dim0 = 64
  g1[4] = 64;                                           // tile_dim1 = 64
  g1[5] = 64;                                           // tensor_dim0_stride
  g1[6] = 0;
  g1[7] = 0;
  i32x4 z4 = {0, 0, 0, 0};
#if defined(__clang_major__) && (__clang_major__ >= 23)
  i32x8 z8 = {0, 0, 0, 0, 0, 0, 0, 0};
  __builtin_amdgcn_tensor_load_to_lds(g0, g1, z4, z4, z8, 0);
#else
  __builtin_amdgcn_tensor_load_to_lds(g0, g1, z4, z4, 0);
#endif
}

// ---------------------------------------------------------------------------
// f32 -> bf16 conversion (vectorized by 4)
// ---------------------------------------------------------------------------
__global__ void cvt_f32_bf16(const float* __restrict__ src, bf16_t* __restrict__ dst, int n) {
  int i = (blockIdx.x * blockDim.x + threadIdx.x) * 4;
  if (i < n) {
    float4 f = *(const float4*)(src + i);
    dst[i + 0] = (bf16_t)f.x;
    dst[i + 1] = (bf16_t)f.y;
    dst[i + 2] = (bf16_t)f.z;
    dst[i + 3] = (bf16_t)f.w;
  }
}

// ---------------------------------------------------------------------------
// Projection GEMM: out[m,n] = sum_k X[m,k] * W[n,k] + bias[n]
//   MODE 0: v   -> bias only, write bf16 TRANSPOSED head-split vT[bh][d][t]
//   MODE 1: k   -> bias + xpos (downscale), bf16 head-split [bh][t][d]
//   MODE 2: q   -> bias, *D^-0.5, xpos (upscale), bf16 head-split [bh][t][d]
//   MODE 3: out -> bias only, write f32 flat [m][n]
// One wave computes a 32(M) x 64(N) strip (2 A-frags share 4 B-frags,
// 8 WMMAs per k-step); block = 4 waves = 128x64 tile.
// ---------------------------------------------------------------------------
template <int MODE>
__global__ __launch_bounds__(128) void proj_gemm(const bf16_t* __restrict__ X,
                                                 const bf16_t* __restrict__ W,
                                                 const float* __restrict__ bias,
                                                 void* __restrict__ out) {
  const int lane = threadIdx.x & 31;
  const int wave = threadIdx.x >> 5;
  const int m0 = blockIdx.y * 128 + wave * 32;
  const int n0 = blockIdx.x * 64;

  v8f acc[2][4];
#pragma unroll
  for (int mi = 0; mi < 2; ++mi)
#pragma unroll
    for (int j = 0; j < 4; ++j)
#pragma unroll
      for (int e = 0; e < 8; ++e) acc[mi][j][e] = 0.0f;

  const bf16_t* xrow = X + (size_t)m0 * EMBED;
  for (int k0 = 0; k0 < EMBED; k0 += 32) {
    if (k0 + 32 < EMBED)
      __builtin_prefetch(xrow + (size_t)(lane & 15) * EMBED + k0 + 32, 0, 0);
    v16bf a0 = load_frag_row(xrow, EMBED, k0);
    v16bf a1 = load_frag_row(xrow + (size_t)16 * EMBED, EMBED, k0);
#pragma unroll
    for (int j = 0; j < 4; ++j) {
      v16bf bfrag = load_frag_row(W + (size_t)(n0 + 16 * j) * EMBED, EMBED, k0);
      acc[0][j] = wmma_bf16(a0, bfrag, acc[0][j]);
      acc[1][j] = wmma_bf16(a1, bfrag, acc[1][j]);
    }
  }

  const int col = lane & 15;
  const int hi  = lane >> 4;

#pragma unroll
  for (int mi = 0; mi < 2; ++mi) {
    const int m0i = m0 + 16 * mi;
    if (MODE == 0) {
      // V: write transposed head-split vT[(bh*64+d)][t], packed v8bf stores.
      const int b = m0i >> 11;
      const int tbase = (m0i & (SEQ - 1)) + 8 * hi;  // 8 consecutive t values
#pragma unroll
      for (int j = 0; j < 4; ++j) {
        const int n = n0 + 16 * j + col;
        const float bval = bias[n];
        const int h = n >> 6, d = n & (HDIM - 1);
        v8bf pack;
#pragma unroll
        for (int r = 0; r < 8; ++r) pack[r] = (bf16_t)(acc[mi][j][r] + bval);
        bf16_t* dst = (bf16_t*)out +
                      ((size_t)((b * HEADS + h) * HDIM + d)) * SEQ + tbase;
        *(v8bf*)dst = pack;
      }
    } else {
#pragma unroll
      for (int j = 0; j < 4; ++j) {
        const int n = n0 + 16 * j + col;
        const float bval = bias[n];
#pragma unroll
        for (int r = 0; r < 8; ++r) {
          const int m = m0i + r + 8 * hi;
          float val = acc[mi][j][r] + bval;
          if (MODE == 3) {
            ((float*)out)[(size_t)m * EMBED + n] = val;
          } else {
            const int b = m >> 11, t = m & (SEQ - 1);
            const int h = n >> 6, d = n & (HDIM - 1);
            if (MODE == 2) val *= 0.125f;  // HEAD_DIM^-0.5
            // xpos: column pair (2j, 2j+1); partner column lives in lane^1.
            const int jh = d >> 1;
            const float base_s =
                (2.0f * (float)jh + 0.4f * (float)HDIM) / (1.4f * (float)HDIM);
            float sc = __expf(((float)t * (1.0f / 1024.0f)) * __logf(base_s));
            if (MODE == 1) sc = 1.0f / sc;  // downscale for k
            const float inv_freq =
                __expf(-((float)jh) * (1.0f / 32.0f) * 9.2103403719761836f);
            const float ang = (float)t * inv_freq;
            float sn, cs;
            __sincosf(ang, &sn, &cs);
            const float partner = __shfl_xor(val, 1, 32);
            const float rot = (d & 1) ? partner : -partner;
            val = val * (cs * sc) + rot * (sn * sc);
            ((bf16_t*)out)[(((size_t)(b * HEADS + h)) * SEQ + t) * HDIM + d] =
                (bf16_t)val;
          }
        }
      }
    }
  }
}

// ---------------------------------------------------------------------------
// Flash attention: one wave per 16-row Q tile of one head-batch.
// s-chunks of 64: K chunks [64 x 64] staged into LDS by the Tensor Data
// Mover (double buffered, s_wait_tensorcnt), 8 score WMMAs + 8 PV WMMAs per
// chunk so softmax/shuffle overhead is amortized over twice the columns.
// V is pre-transposed so PV B-fragments are contiguous loads.
// ---------------------------------------------------------------------------
__global__ __launch_bounds__(128) void attn_flash(const bf16_t* __restrict__ q,
                                                  const bf16_t* __restrict__ k,
                                                  const bf16_t* __restrict__ vT,
                                                  bf16_t* __restrict__ outm) {
  __shared__ __align__(16) bf16_t lds_k[4][2][64 * HDIM];  // 4 waves x 2 bufs
  __shared__ __align__(16) bf16_t lds_p[4][16 * 64];

  const int lane = threadIdx.x & 31;
  const int wave = threadIdx.x >> 5;
  const int tile = blockIdx.x * 4 + wave;
  const int bh = tile >> 7;           // 128 Q-tiles per head-batch
  const int t0 = (tile & 127) * 16;
  const int b = bh >> 4, h = bh & 15;

  const bf16_t* qh  = q  + (size_t)bh * SEQ * HDIM;
  const bf16_t* kh  = k  + (size_t)bh * SEQ * HDIM;
  const bf16_t* vTh = vT + (size_t)bh * HDIM * SEQ;

  const v16bf aq0 = load_frag_row(qh + (size_t)t0 * HDIM, HDIM, 0);
  const v16bf aq1 = load_frag_row(qh + (size_t)t0 * HDIM, HDIM, 32);

  v8f oacc[4];
#pragma unroll
  for (int j = 0; j < 4; ++j)
#pragma unroll
    for (int e = 0; e < 8; ++e) oacc[j][e] = 0.0f;
  float mrow[8], lrow[8];
#pragma unroll
  for (int r = 0; r < 8; ++r) { mrow[r] = -3.0e38f; lrow[r] = 0.0f; }

  bf16_t* pl = &lds_p[wave][0];
  const int col = lane & 15;
  const int hi  = lane >> 4;

  const unsigned lds_k_addr0 = (unsigned)(uintptr_t)(&lds_k[wave][0][0]);
  const unsigned lds_k_addr1 = (unsigned)(uintptr_t)(&lds_k[wave][1][0]);
  const unsigned long long kbase = (unsigned long long)(uintptr_t)kh;

  // Prime buffer 0 with the first K chunk via the TDM.
  tdm_load_k64x64(lds_k_addr0, kbase);

  for (int s0 = 0; s0 < SEQ; s0 += 64) {
    const int buf = (s0 >> 6) & 1;
    if (s0 + 64 < SEQ) {
      // Kick the next chunk into the other buffer, then wait until only
      // that one is outstanding (i.e. the current chunk has landed).
      tdm_load_k64x64(buf ? lds_k_addr0 : lds_k_addr1,
                      kbase + (unsigned long long)(s0 + 64) * HDIM * 2u);
      __builtin_amdgcn_s_wait_tensorcnt(1);
    } else {
      __builtin_amdgcn_s_wait_tensorcnt(0);
    }
    const bf16_t* kcur = &lds_k[wave][buf][0];

    // ---- scores S = Q * K^T for a 16x64 chunk (B[d][s] = K[s][d]) ---------
    v8f sc[4];
#pragma unroll
    for (int j = 0; j < 4; ++j) {
#pragma unroll
      for (int e = 0; e < 8; ++e) sc[j][e] = 0.0f;
      v16bf b0 = load_frag_row(kcur + (size_t)(16 * j) * HDIM, HDIM, 0);
      v16bf b1 = load_frag_row(kcur + (size_t)(16 * j) * HDIM, HDIM, 32);
      sc[j] = wmma_bf16(aq0, b0, sc[j]);
      sc[j] = wmma_bf16(aq1, b1, sc[j]);
    }

    // ---- online softmax over the 64-column chunk --------------------------
#pragma unroll
    for (int r = 0; r < 8; ++r) {
      const float v0 = sc[0][r], v1 = sc[1][r];
      const float v2 = sc[2][r], v3 = sc[3][r];
      float cmax = fmaxf(fmaxf(v0, v1), fmaxf(v2, v3));
      cmax = fmaxf(cmax, __shfl_xor(cmax, 1, 32));
      cmax = fmaxf(cmax, __shfl_xor(cmax, 2, 32));
      cmax = fmaxf(cmax, __shfl_xor(cmax, 4, 32));
      cmax = fmaxf(cmax, __shfl_xor(cmax, 8, 32));
      const float nm = fmaxf(mrow[r], cmax);
      const float corr = __expf(mrow[r] - nm);
      const float p0 = __expf(v0 - nm);
      const float p1 = __expf(v1 - nm);
      const float p2 = __expf(v2 - nm);
      const float p3 = __expf(v3 - nm);
      float psum = (p0 + p1) + (p2 + p3);
      psum += __shfl_xor(psum, 1, 32);
      psum += __shfl_xor(psum, 2, 32);
      psum += __shfl_xor(psum, 4, 32);
      psum += __shfl_xor(psum, 8, 32);
      lrow[r] = lrow[r] * corr + psum;
      mrow[r] = nm;
      oacc[0][r] *= corr; oacc[1][r] *= corr;
      oacc[2][r] *= corr; oacc[3][r] *= corr;
      const int row = r + 8 * hi;
      pl[row * 64 + col]      = (bf16_t)p0;
      pl[row * 64 + 16 + col] = (bf16_t)p1;
      pl[row * 64 + 32 + col] = (bf16_t)p2;
      pl[row * 64 + 48 + col] = (bf16_t)p3;
    }

    // ---- O += P * V: A-frags of P from LDS; V B-frags contiguous from vT --
    {
      v16bf ap0 = load_frag_row(pl, 64, 0);
      v16bf ap1 = load_frag_row(pl, 64, 32);
#pragma unroll
      for (int j = 0; j < 4; ++j) {
        v16bf bv0 = load_frag_row(vTh + (size_t)(16 * j) * SEQ, SEQ, s0);
        v16bf bv1 = load_frag_row(vTh + (size_t)(16 * j) * SEQ, SEQ, s0 + 32);
        oacc[j] = wmma_bf16(ap0, bv0, oacc[j]);
        oacc[j] = wmma_bf16(ap1, bv1, oacc[j]);
      }
    }
  }

  // ---- normalize and write merged [b][t][h*64+d] bf16 ----------------------
#pragma unroll
  for (int r = 0; r < 8; ++r) {
    const float inv = 1.0f / lrow[r];
    const int t = t0 + r + 8 * hi;
    const size_t rowoff = ((size_t)b * SEQ + t) * EMBED + h * HDIM;
#pragma unroll
    for (int j = 0; j < 4; ++j)
      outm[rowoff + 16 * j + col] = (bf16_t)(oacc[j][r] * inv);
  }
}

// ---------------------------------------------------------------------------
// Host launcher
// ---------------------------------------------------------------------------
extern "C" void kernel_launch(void* const* d_in, const int* in_sizes, int n_in,
                              void* d_out, int out_size, void* d_ws, size_t ws_size,
                              hipStream_t stream) {
  const float* q_in = (const float*)d_in[0];
  const float* k_in = (const float*)d_in[1];
  const float* v_in = (const float*)d_in[2];
  const float* Wq = (const float*)d_in[3];
  const float* bq = (const float*)d_in[4];
  const float* Wk = (const float*)d_in[5];
  const float* bk = (const float*)d_in[6];
  const float* Wv = (const float*)d_in[7];
  const float* bv = (const float*)d_in[8];
  const float* Wo = (const float*)d_in[9];
  const float* bo = (const float*)d_in[10];

  char* ws = (char*)d_ws;
  const size_t MB = 1024ull * 1024ull;
  bf16_t* xq  = (bf16_t*)(ws + 0 * MB);   // 8 MB each activation (bf16)
  bf16_t* xk  = (bf16_t*)(ws + 8 * MB);
  bf16_t* xv  = (bf16_t*)(ws + 16 * MB);
  bf16_t* wqb = (bf16_t*)(ws + 24 * MB);  // 2 MB each weight (bf16)
  bf16_t* wkb = (bf16_t*)(ws + 26 * MB);
  bf16_t* wvb = (bf16_t*)(ws + 28 * MB);
  bf16_t* wob = (bf16_t*)(ws + 30 * MB);
  bf16_t* qhs = (bf16_t*)(ws + 32 * MB);  // head-split [32][2048][64] bf16
  bf16_t* khs = (bf16_t*)(ws + 40 * MB);
  bf16_t* vts = (bf16_t*)(ws + 48 * MB);  // V transposed [32][64][2048] bf16
  bf16_t* am  = (bf16_t*)(ws + 56 * MB);  // merged attention output bf16

  const int nX = MTOT * EMBED;   // 4194304
  const int nW = EMBED * EMBED;  // 1048576

  cvt_f32_bf16<<<nX / 1024, 256, 0, stream>>>(q_in, xq, nX);
  cvt_f32_bf16<<<nX / 1024, 256, 0, stream>>>(k_in, xk, nX);
  cvt_f32_bf16<<<nX / 1024, 256, 0, stream>>>(v_in, xv, nX);
  cvt_f32_bf16<<<nW / 1024, 256, 0, stream>>>(Wq, wqb, nW);
  cvt_f32_bf16<<<nW / 1024, 256, 0, stream>>>(Wk, wkb, nW);
  cvt_f32_bf16<<<nW / 1024, 256, 0, stream>>>(Wv, wvb, nW);
  cvt_f32_bf16<<<nW / 1024, 256, 0, stream>>>(Wo, wob, nW);

  dim3 gg(EMBED / 64, MTOT / 128);  // (16, 32), 128 threads = 4 waves
  proj_gemm<2><<<gg, 128, 0, stream>>>(xq, wqb, bq, (void*)qhs);
  proj_gemm<1><<<gg, 128, 0, stream>>>(xk, wkb, bk, (void*)khs);
  proj_gemm<0><<<gg, 128, 0, stream>>>(xv, wvb, bv, (void*)vts);

  attn_flash<<<(NBH * SEQ / 16) / 4, 128, 0, stream>>>(qhs, khs, vts, am);

  proj_gemm<3><<<gg, 128, 0, stream>>>(am, wob, bo, d_out);
}
